// Memory_7378753815338
// MI455X (gfx1250) — matrix operand order repaired and checked by reading, not compiled, not run on hardware
//
#include <hip/hip_runtime.h>
#include <hip/hip_bf16.h>
#include <hip/hip_fp16.h>

typedef __attribute__((ext_vector_type(16))) _Float16 v16h;
typedef __attribute__((ext_vector_type(8)))  float    v8f;

#define B_ROWS   512
#define DIM      256
#define M_KEYS   262144
#define TOPK     256
#define NBINS    512
#define CAND_MAX 1024
#define MARGIN_F 0.1f

// ---------- helpers ----------
__device__ __forceinline__ unsigned sortable_u32(float f) {
    unsigned u = __float_as_uint(f);
    return (u & 0x80000000u) ? ~u : (u | 0x80000000u);
}
__device__ __forceinline__ unsigned score_bucket(float f) {
    return sortable_u32(f) >> 23;   // top 9 bits -> 512 bins, monotonic in f
}

// A fragment: q tile 16x32 f16. Lane L<16: M=L, K=kb..kb+15;
// lane L>=16: M=L-16, K=kb+16..kb+31.
__device__ __forceinline__ v16h load_a_frag(const _Float16* __restrict__ qf16,
                                            int rowBase, int kBase, int lane) {
    int m = rowBase + (lane & 15);
    const _Float16* p = qf16 + m * DIM + kBase + ((lane >> 4) << 4);
    v16h a;
#pragma unroll
    for (int i = 0; i < 16; ++i) a[i] = p[i];
    return a;
}

// B fragment: keys.T tile 32x16 f16, converted from f32 in-register.
// Lanes 0-15 hold K=kb..kb+15 of column N=lane; lanes 16-31 hold K=kb+16..kb+31.
// Each lane reads 64 contiguous bytes of one key row (coalesced HBM stream).
__device__ __forceinline__ v16h load_b_frag(const float* __restrict__ keys,
                                            int colBase, int kBase, int lane) {
    int n = colBase + (lane & 15);
    const float* p = keys + (size_t)n * DIM + kBase + ((lane >> 4) << 4);
    __builtin_prefetch((const void*)(p + 32), 0, 1);   // global_prefetch_b8 on next K-chunk
    v16h b;
#pragma unroll
    for (int i = 0; i < 16; ++i) b[i] = (_Float16)p[i];
    return b;
}

// ---------- 0: zero workspace state ----------
__global__ void k_init(unsigned* __restrict__ ghist, unsigned* __restrict__ ccount,
                       float* __restrict__ lossp) {
    int i = blockIdx.x * blockDim.x + threadIdx.x;
    if (i < B_ROWS * NBINS) ghist[i] = 0u;
    if (i < B_ROWS) ccount[i] = 0u;
    if (i == 0) *lossp = 0.0f;
}

// ---------- 1: q = normalize(x @ W.T + b) -> f16 ----------
__global__ void k_compute_q(const float* __restrict__ x, const float* __restrict__ W,
                            const float* __restrict__ b, _Float16* __restrict__ qf16) {
    __shared__ float xr[DIM];
    __shared__ float red[256];
    const int row = blockIdx.x;
    const int t   = threadIdx.x;
    xr[t] = x[row * DIM + t];
    __syncthreads();
    float acc = b[t];
    const float* wr = W + t * DIM;
#pragma unroll 4
    for (int k = 0; k < DIM; ++k) acc = fmaf(xr[k], wr[k], acc);
    red[t] = acc * acc;
    __syncthreads();
    for (int s = 128; s > 0; s >>= 1) { if (t < s) red[t] += red[t + s]; __syncthreads(); }
    const float rn = rsqrtf(red[0] + 1e-30f);
    qf16[row * DIM + t] = (_Float16)(acc * rn);
}

// ---------- 2: GEMM (WMMA f16) + per-row score histogram ----------
// grid = (M/1024, 32), block = 128 (4 waves). Each wave: 8 pairs of col-tiles
// (dual accumulators -> independent WMMA chains fill the 4-coexec WAR hazard
// slots that otherwise become v_nops). Block covers 16 rows x 1024 cols.
__global__ void __launch_bounds__(128)
k_gemm_hist(const _Float16* __restrict__ qf16, const float* __restrict__ keys,
            unsigned* __restrict__ ghist) {
    __shared__ unsigned sh[16 * NBINS];          // 32 KB
    for (int i = threadIdx.x; i < 16 * NBINS; i += blockDim.x) sh[i] = 0u;
    __syncthreads();

    const int lane    = threadIdx.x & 31;
    const int wave    = threadIdx.x >> 5;
    const int rowBase = blockIdx.y * 16;
    const int rsub    = (lane >> 4) << 3;

    // q fragments for this 16-row block: resident in 64 VGPRs for the kernel
    v16h Afr[8];
#pragma unroll
    for (int k8 = 0; k8 < 8; ++k8) Afr[k8] = load_a_frag(qf16, rowBase, k8 * 32, lane);

    for (int tt = 0; tt < 8; ++tt) {
        const int colBase0 = (blockIdx.x * 64 + wave * 16 + tt * 2) * 16;
        const int colBase1 = colBase0 + 16;
        v8f acc0 = {0.f, 0.f, 0.f, 0.f, 0.f, 0.f, 0.f, 0.f};
        v8f acc1 = {0.f, 0.f, 0.f, 0.f, 0.f, 0.f, 0.f, 0.f};
#pragma unroll
        for (int k8 = 0; k8 < 8; ++k8) {
            v16h b0 = load_b_frag(keys, colBase0, k8 * 32, lane);
            v16h b1 = load_b_frag(keys, colBase1, k8 * 32, lane);
            acc0 = __builtin_amdgcn_wmma_f32_16x16x32_f16(
                false, Afr[k8], false, b0, (short)0, acc0, false, false);
            acc1 = __builtin_amdgcn_wmma_f32_16x16x32_f16(
                false, Afr[k8], false, b1, (short)0, acc1, false, false);
        }
#pragma unroll
        for (int r = 0; r < 8; ++r) {
            atomicAdd(&sh[(r + rsub) * NBINS + score_bucket(acc0[r])], 1u);
            atomicAdd(&sh[(r + rsub) * NBINS + score_bucket(acc1[r])], 1u);
        }
    }
    __syncthreads();
    for (int i = threadIdx.x; i < 16 * NBINS; i += blockDim.x) {
        const unsigned v = sh[i];
        if (v) atomicAdd(&ghist[(rowBase + (i >> 9)) * NBINS + (i & (NBINS - 1))], v);
    }
}

// ---------- 3: per-row threshold bucket (suffix scan from the top) ----------
__global__ void k_threshold(const unsigned* __restrict__ ghist, int* __restrict__ tb) {
    const int row = blockIdx.x * blockDim.x + threadIdx.x;
    if (row >= B_ROWS) return;
    const unsigned* h = ghist + row * NBINS;
    unsigned cum = 0; int t = 0;
    for (int bin = NBINS - 1; bin >= 0; --bin) {
        cum += h[bin];
        if (cum >= TOPK) { t = bin; break; }
    }
    tb[row] = t;
}

// ---------- 4: GEMM again, compact candidates >= threshold ----------
__global__ void __launch_bounds__(128)
k_gemm_compact(const _Float16* __restrict__ qf16, const float* __restrict__ keys,
               const int* __restrict__ tb, unsigned* __restrict__ ccount,
               float* __restrict__ cscore, int* __restrict__ cidx) {
    const int lane    = threadIdx.x & 31;
    const int wave    = threadIdx.x >> 5;
    const int rowBase = blockIdx.y * 16;
    const int rsub    = (lane >> 4) << 3;
    const int col0    = lane & 15;

    v16h Afr[8];
#pragma unroll
    for (int k8 = 0; k8 < 8; ++k8) Afr[k8] = load_a_frag(qf16, rowBase, k8 * 32, lane);

    // thresholds for the 8 rows this lane produces
    int tbl[8];
#pragma unroll
    for (int r = 0; r < 8; ++r) tbl[r] = tb[rowBase + r + rsub];

    for (int tt = 0; tt < 8; ++tt) {
        const int colBase0 = (blockIdx.x * 64 + wave * 16 + tt * 2) * 16;
        const int colBase1 = colBase0 + 16;
        v8f acc0 = {0.f, 0.f, 0.f, 0.f, 0.f, 0.f, 0.f, 0.f};
        v8f acc1 = {0.f, 0.f, 0.f, 0.f, 0.f, 0.f, 0.f, 0.f};
#pragma unroll
        for (int k8 = 0; k8 < 8; ++k8) {
            v16h b0 = load_b_frag(keys, colBase0, k8 * 32, lane);
            v16h b1 = load_b_frag(keys, colBase1, k8 * 32, lane);
            acc0 = __builtin_amdgcn_wmma_f32_16x16x32_f16(
                false, Afr[k8], false, b0, (short)0, acc0, false, false);
            acc1 = __builtin_amdgcn_wmma_f32_16x16x32_f16(
                false, Afr[k8], false, b1, (short)0, acc1, false, false);
        }
#pragma unroll
        for (int r = 0; r < 8; ++r) {
            const int row = rowBase + r + rsub;
            const float s0 = acc0[r];
            if ((int)score_bucket(s0) >= tbl[r]) {
                const unsigned pos = atomicAdd(&ccount[row], 1u);
                if (pos < CAND_MAX) {
                    cscore[row * CAND_MAX + pos] = s0;
                    cidx[row * CAND_MAX + pos]   = colBase0 + col0;
                }
            }
            const float s1 = acc1[r];
            if ((int)score_bucket(s1) >= tbl[r]) {
                const unsigned pos = atomicAdd(&ccount[row], 1u);
                if (pos < CAND_MAX) {
                    cscore[row * CAND_MAX + pos] = s1;
                    cidx[row * CAND_MAX + pos]   = colBase1 + col0;
                }
            }
        }
    }
}

// ---------- 5: per-row sort, softmax, y_hat, margin loss ----------
__global__ void __launch_bounds__(256)
k_finalize(const unsigned* __restrict__ ccount, const float* __restrict__ cscore,
           const int* __restrict__ cidx, const int* __restrict__ values,
           const int* __restrict__ y, float* __restrict__ out) {
    __shared__ float s[CAND_MAX];
    __shared__ int   id[CAND_MAX];
    __shared__ float red[256];
    __shared__ float rp[256];
    __shared__ float rn[256];
    __shared__ int   rh[256];

    const int row = blockIdx.x;
    const int t   = threadIdx.x;
    const int n   = (int)min(ccount[row], (unsigned)CAND_MAX);
    const float NEG_INF = -__builtin_inff();

    for (int i = t; i < CAND_MAX; i += 256) {
        if (i < n) { s[i] = cscore[row * CAND_MAX + i]; id[i] = cidx[row * CAND_MAX + i]; }
        else       { s[i] = NEG_INF;                    id[i] = -1; }
    }
    __syncthreads();

    // bitonic sort, descending
    for (int k2 = 2; k2 <= CAND_MAX; k2 <<= 1) {
        for (int j = k2 >> 1; j > 0; j >>= 1) {
            for (int i = t; i < CAND_MAX; i += 256) {
                const int ixj = i ^ j;
                if (ixj > i) {
                    const bool asc = ((i & k2) != 0);
                    const float si = s[i], sj = s[ixj];
                    if ((si > sj) == asc) {
                        s[i] = sj; s[ixj] = si;
                        const int tmp = id[i]; id[i] = id[ixj]; id[ixj] = tmp;
                    }
                }
            }
            __syncthreads();
        }
    }

    // softmax over top-256 (TEMP = 1.0)
    const float maxv = s[0];
    const float sc   = (t < TOPK) ? s[t] : NEG_INF;
    const float e    = (sc == NEG_INF) ? 0.0f : __expf(sc - maxv);
    red[t] = e;
    __syncthreads();
    for (int st = 128; st > 0; st >>= 1) { if (t < st) red[t] += red[t + st]; __syncthreads(); }
    const float denom = red[0] + 1e-30f;
    if (t < TOPK) out[B_ROWS + row * TOPK + t] = e / denom;
    if (t == 0)   out[row] = (id[0] >= 0) ? (float)values[id[0]] : 0.0f;

    // margin loss terms
    const int yr = y[row];
    float posv = 0.0f, negv = 0.0f; int hp = 0;
    if (t < TOPK && id[t] >= 0) {
        const bool c = (values[id[t]] == yr);
        if (c) { posv = sc; hp = 1; } else { negv = sc; }
    }
    rp[t] = posv; rn[t] = negv; rh[t] = hp;
    __syncthreads();
    for (int st = 128; st > 0; st >>= 1) {
        if (t < st) {
            rp[t] = fmaxf(rp[t], rp[t + st]);
            rn[t] = fmaxf(rn[t], rn[t + st]);
            rh[t] = rh[t] | rh[t + st];
        }
        __syncthreads();
    }
    if (t == 0) {
        const float pos = rh[0] ? rp[0] : 0.0f;
        const float row_loss = fmaxf(rn[0] - pos + MARGIN_F, 0.0f);
        atomicAdd(&out[B_ROWS + B_ROWS * TOPK], row_loss * (1.0f / (float)B_ROWS));
    }
}

// ---------- launch ----------
extern "C" void kernel_launch(void* const* d_in, const int* in_sizes, int n_in,
                              void* d_out, int out_size, void* d_ws, size_t ws_size,
                              hipStream_t stream) {
    const float* x      = (const float*)d_in[0];
    const int*   y      = (const int*)  d_in[1];
    const float* keys   = (const float*)d_in[2];
    const int*   values = (const int*)  d_in[3];
    const float* W      = (const float*)d_in[4];
    const float* b      = (const float*)d_in[5];
    float*       out    = (float*)d_out;

    // workspace layout (bytes)
    char* ws = (char*)d_ws;
    _Float16* qf16   = (_Float16*)(ws + 0);                         // 256 KB
    unsigned* ghist  = (unsigned*)(ws + 262144);                    // 1 MB
    int*      tb     = (int*)     (ws + 1310720);                   // 2 KB
    unsigned* ccount = (unsigned*)(ws + 1312768);                   // 2 KB
    float*    cscore = (float*)   (ws + 1314816);                   // 2 MB
    int*      cidx   = (int*)     (ws + 3411968);                   // 2 MB

    float* lossp = out + B_ROWS + B_ROWS * TOPK;

    k_init<<<(B_ROWS * NBINS + 255) / 256, 256, 0, stream>>>(ghist, ccount, lossp);
    k_compute_q<<<B_ROWS, 256, 0, stream>>>(x, W, b, qf16);

    dim3 ggrid(M_KEYS / 1024, B_ROWS / 16);   // (256, 32)
    k_gemm_hist<<<ggrid, 128, 0, stream>>>(qf16, keys, ghist);
    k_threshold<<<(B_ROWS + 255) / 256, 256, 0, stream>>>(ghist, tb);
    k_gemm_compact<<<ggrid, 128, 0, stream>>>(qf16, keys, tb, ccount, cscore, cidx);
    k_finalize<<<B_ROWS, 256, 0, stream>>>(ccount, cscore, cidx, values, y, out);
}
